// SelfInteraction_71554155151993
// MI455X (gfx1250) — matrix operand order repaired and checked by reading, not compiled, not run on hardware
//
#include <hip/hip_runtime.h>
#include <math.h>

typedef __attribute__((ext_vector_type(16))) _Float16 v16h;
typedef __attribute__((ext_vector_type(8)))  _Float16 v8h;
typedef __attribute__((ext_vector_type(8)))  float    v8f;

#define WAVES 2
#define RPW 16                 // rows per wave (WMMA M)
#define RPB (WAVES * RPW)      // rows per block
#define ALD 648                // padded LDS stride for A tiles (halves)

// packed-weight geometry
#define KT_S 20                // 640/32 k-tiles for W_s
#define NT_S 8                 // 128/16 n-tiles for W_s
#define KT_V 18                // 576/32 k-tiles for W_v
#define NT_V 4                 // 64/16  n-tiles for W_v
#define WS_ELEMS (640 * 128)
#define WV_ELEMS (576 * 64)
#define PACKED_BYTES ((WS_ELEMS + WV_ELEMS) * 2)

// triu_indices(8): 36 (i,j) pairs, i<=j
__device__ __constant__ unsigned char cIU[36] = {
  0,0,0,0,0,0,0,0, 1,1,1,1,1,1,1, 2,2,2,2,2,2, 3,3,3,3,3, 4,4,4,4, 5,5,5, 6,6, 7};
__device__ __constant__ unsigned char cJU[36] = {
  0,1,2,3,4,5,6,7, 1,2,3,4,5,6,7, 2,3,4,5,6,7, 3,4,5,6,7, 4,5,6,7, 5,6,7, 6,7, 7};

// fast sigmoid: v_exp_f32 + v_rcp_f32 (avoids the precise-division sequence)
__device__ __forceinline__ float sigm(float x) {
  return __builtin_amdgcn_rcpf(1.0f + __expf(-x));
}

// ---------------------------------------------------------------------------
// Setup kernel: convert W_s / W_v to f16 (scale folded) in WMMA-fragment order.
// Fragment (kt, nt): 32 lanes x 16 contiguous halves; lane = hi*16 + ncol where
// hi = (k%32)>>4, j = k%16 (B layout: lanes 0-15 hold K 0..15, 16-31 hold 16..31).
// ---------------------------------------------------------------------------
__global__ void pack_weights_kernel(const float* __restrict__ Ws,
                                    const float* __restrict__ Wv,
                                    _Float16* __restrict__ WpS,
                                    _Float16* __restrict__ WpV)
{
  const float wsS = 0.039528470752104741f;   // 1/sqrt(640)
  const float wsV = 0.041666666666666664f;   // 1/sqrt(576)
  int idx = blockIdx.x * blockDim.x + threadIdx.x;
  if (idx < WS_ELEMS) {
    const int k = idx >> 7, nn = idx & 127;
    const int kt = k >> 5, rem = k & 31, hi = rem >> 4, j = rem & 15;
    const int nt = nn >> 4, ncol = nn & 15, lane = hi * 16 + ncol;
    WpS[((size_t)(kt * NT_S + nt) * 32 + lane) * 16 + j] = (_Float16)(Ws[idx] * wsS);
  } else if (idx < WS_ELEMS + WV_ELEMS) {
    const int e = idx - WS_ELEMS;
    const int k = e >> 6, nn = e & 63;
    const int kt = k >> 5, rem = k & 31, hi = rem >> 4, j = rem & 15;
    const int nt = nn >> 4, ncol = nn & 15, lane = hi * 16 + ncol;
    WpV[((size_t)(kt * NT_V + nt) * 32 + lane) * 16 + j] = (_Float16)(Wv[e] * wsV);
  }
}

// load one A fragment (16x32 f16) for this lane: two ds_load_b128's
__device__ __forceinline__ v16h load_a_frag(const _Float16* Ar) {
  const v8h* Ap = (const v8h*)Ar;          // 16-byte aligned
  v8h lo = Ap[0];
  v8h hi8 = Ap[1];                          // +8 halves
  return __builtin_shufflevector(lo, hi8, 0,1,2,3,4,5,6,7,8,9,10,11,12,13,14,15);
}

// scal epilogue for one accumulator / one column
__device__ __forceinline__ void scal_epi(const v8f& acc, int col, int row0, int omBase,
                                         const float* __restrict__ cnoise,
                                         const float* __restrict__ nw,
                                         const float* __restrict__ nb,
                                         float* __restrict__ outf,
                                         float (*G)[64])
{
  #pragma unroll
  for (int r = 0; r < 8; ++r) {
    const int mm  = r + omBase;
    const int row = row0 + mm;
    const float x = acc[r];
    if (col < 64) {
      const float sc  = x * sigm(x);                                // silu
      const float scl = 1.0f + cnoise[row] * nw[col] + nb[col];
      outf[(size_t)row * 256 + col] = sc * scl;
    } else {
      G[mm][col - 64] = sigm(x);                                    // gate
    }
  }
}

__device__ __forceinline__ void vec_epi(const v8f& acc, int o, int c, int row0, int omBase,
                                        const float* __restrict__ cnoise,
                                        const float* __restrict__ nw,
                                        const float* __restrict__ nb,
                                        float* __restrict__ outf,
                                        const float (*G)[64])
{
  #pragma unroll
  for (int r = 0; r < 8; ++r) {
    const int mm  = r + omBase;
    const int row = row0 + mm;
    const float x = acc[r];
    const float g   = G[mm][o];
    const float scl = 1.0f + cnoise[row] * nw[64 + o] + nb[64 + o];
    outf[(size_t)row * 256 + 64 + o * 3 + c] = x * g * scl;
  }
}

template <bool PACKED>
__global__ __launch_bounds__(64)
void si_fused_kernel(const float* __restrict__ F,      // (n,256)
                     const float* __restrict__ coords, // (n,3)
                     const float* __restrict__ cnoise, // (n,)
                     const float* __restrict__ Ws,     // (640,128)
                     const float* __restrict__ Wv,     // (576,64)
                     const _Float16* __restrict__ WpS, // packed f16
                     const _Float16* __restrict__ WpV, // packed f16
                     const float* __restrict__ nw,     // (128,)
                     const float* __restrict__ nb,     // (128,)
                     float* __restrict__ outc,         // (n,3)
                     float* __restrict__ outf,         // (n,256)
                     int n)
{
  __shared__ alignas(16) _Float16 Alds[WAVES][RPW][ALD];
  __shared__ float Glds[WAVES][RPW][64];

  const int wave = threadIdx.x >> 5;
  const int lane = threadIdx.x & 31;
  const int row0 = blockIdx.x * RPB + wave * RPW;

  // ---- coords passthrough ----
  {
    const int base = blockIdx.x * RPB * 3;
    for (int t = threadIdx.x; t < RPB * 3; t += blockDim.x)
      outc[base + t] = coords[base + t];
  }

  const float SQRT2  = 1.41421356237309515f;
  const float INVSQ3 = 0.57735026918962584f;

  // ---- Phase 0: build scal_in (16 x 640) tile in LDS as f16 ----
  for (int t = lane; t < RPW * 640; t += 32) {
    const int r = t / 640, k = t - r * 640;
    const float* f = F + (size_t)(row0 + r) * 256;
    float val;
    if (k < 64) {
      val = f[k];
    } else if (k < 352) {                       // ss: s_i * s_j * pf
      const int idx = k - 64, fx = idx / 36, p = idx - fx * 36;
      const int i = cIU[p], j = cJU[p];
      const float pf = (i == j) ? 1.0f : SQRT2;
      val = f[fx * 8 + i] * f[fx * 8 + j] * pf;
    } else {                                    // vv0: dot(v_i, v_j) * pf / sqrt(3)
      const int idx = k - 352, fx = idx / 36, p = idx - fx * 36;
      const int i = cIU[p], j = cJU[p];
      const float* vf = f + 64 + fx * 24;
      const float d = vf[i*3]*vf[j*3] + vf[i*3+1]*vf[j*3+1] + vf[i*3+2]*vf[j*3+2];
      const float pf = (i == j) ? 1.0f : SQRT2;
      val = d * pf * INVSQ3;
    }
    Alds[wave][r][k] = (_Float16)val;
  }
  __syncthreads();

  // WMMA f16 operand lane mappings (contiguous k order, same for A and B)
  const int hi     = lane >> 4;
  const int mrow   = lane & 15;          // A-matrix M row fed by this lane
  const int kbA    = hi * 16;            // A: lanes 0-15 -> K 0..15; 16-31 -> K 16..31
  const int kbB    = hi * 16;            // B: lanes 0-15 -> K 0..15; 16-31 -> K 16..31
  const int ncol   = lane & 15;
  const int omBase = hi ? 8 : 0;

  const float wsS = 0.039528470752104741f;
  const float wsV = 0.041666666666666664f;

  // ---- Phase 1: scal_mix = scal_in @ Ws (2 N-tiles per pass) ----
  for (int nt = 0; nt < NT_S; nt += 2) {
    const int n0 = nt * 16;
    v8f acc0 = {}, acc1 = {};
    for (int k0 = 0; k0 < 640; k0 += 32) {
      const v16h a = load_a_frag(&Alds[wave][mrow][k0 + kbA]);
      v16h b0, b1;
      if (PACKED) {
        const int kt = k0 >> 5;
        b0 = *(const v16h*)(WpS + ((size_t)(kt * NT_S + nt    ) * 32 + lane) * 16);
        b1 = *(const v16h*)(WpS + ((size_t)(kt * NT_S + nt + 1) * 32 + lane) * 16);
      } else {
        const float* Wc = Ws + (size_t)(k0 + kbB) * 128 + n0 + ncol;
        #pragma unroll
        for (int j = 0; j < 16; ++j) {
          b0[j] = (_Float16)(Wc[j * 128]      * wsS);
          b1[j] = (_Float16)(Wc[j * 128 + 16] * wsS);
        }
      }
      acc0 = __builtin_amdgcn_wmma_f32_16x16x32_f16(false, a, false, b0,
                                                    (short)0, acc0, false, false);
      acc1 = __builtin_amdgcn_wmma_f32_16x16x32_f16(false, a, false, b1,
                                                    (short)0, acc1, false, false);
    }
    scal_epi(acc0, n0 + ncol,      row0, omBase, cnoise, nw, nb, outf, Glds[wave]);
    scal_epi(acc1, n0 + 16 + ncol, row0, omBase, cnoise, nw, nb, outf, Glds[wave]);
  }
  __syncthreads();

  // ---- Phase 2: per channel c, vec_mix = vec_in[:,:,c] @ Wv (2 N-tiles/pass) ----
  for (int c = 0; c < 3; ++c) {
    for (int t = lane; t < RPW * 576; t += 32) {
      const int r = t / 576, k = t - r * 576;
      const float* f = F + (size_t)(row0 + r) * 256;
      float val;
      if (k < 64) {                                     // v[f][j][c], f*8+j == k
        val = f[64 + k * 3 + c];
      } else {                                          // sv = s[f][i] * v[f][j][c]
        const int idx = k - 64;
        const int fx = idx >> 6, i = (idx >> 3) & 7, j = idx & 7;
        val = f[fx * 8 + i] * f[64 + (fx * 8 + j) * 3 + c];
      }
      Alds[wave][r][k] = (_Float16)val;
    }
    __syncthreads();

    for (int nt = 0; nt < NT_V; nt += 2) {
      const int n0 = nt * 16;
      v8f acc0 = {}, acc1 = {};
      for (int k0 = 0; k0 < 576; k0 += 32) {
        const v16h a = load_a_frag(&Alds[wave][mrow][k0 + kbA]);
        v16h b0, b1;
        if (PACKED) {
          const int kt = k0 >> 5;
          b0 = *(const v16h*)(WpV + ((size_t)(kt * NT_V + nt    ) * 32 + lane) * 16);
          b1 = *(const v16h*)(WpV + ((size_t)(kt * NT_V + nt + 1) * 32 + lane) * 16);
        } else {
          const float* Wc = Wv + (size_t)(k0 + kbB) * 64 + n0 + ncol;
          #pragma unroll
          for (int j = 0; j < 16; ++j) {
            b0[j] = (_Float16)(Wc[j * 64]      * wsV);
            b1[j] = (_Float16)(Wc[j * 64 + 16] * wsV);
          }
        }
        acc0 = __builtin_amdgcn_wmma_f32_16x16x32_f16(false, a, false, b0,
                                                      (short)0, acc0, false, false);
        acc1 = __builtin_amdgcn_wmma_f32_16x16x32_f16(false, a, false, b1,
                                                      (short)0, acc1, false, false);
      }
      vec_epi(acc0, n0 + ncol,      c, row0, omBase, cnoise, nw, nb, outf, Glds[wave]);
      vec_epi(acc1, n0 + 16 + ncol, c, row0, omBase, cnoise, nw, nb, outf, Glds[wave]);
    }
    __syncthreads();
  }
}

extern "C" void kernel_launch(void* const* d_in, const int* in_sizes, int n_in,
                              void* d_out, int out_size, void* d_ws, size_t ws_size,
                              hipStream_t stream) {
  const float* features = (const float*)d_in[0];
  const float* coords   = (const float*)d_in[1];
  const float* c_noise  = (const float*)d_in[2];
  const float* W_s      = (const float*)d_in[3];
  const float* W_v      = (const float*)d_in[4];
  const float* noise_w  = (const float*)d_in[5];
  const float* noise_b  = (const float*)d_in[6];

  const int n = in_sizes[0] / 256;          // 32768
  float* outc = (float*)d_out;              // (n,3)
  float* outf = outc + (size_t)n * 3;       // (n,256)

  const int blocks = n / RPB;               // 1024 blocks of 64 threads (2 waves)

  if (ws_size >= (size_t)PACKED_BYTES) {
    _Float16* WpS = (_Float16*)d_ws;
    _Float16* WpV = WpS + WS_ELEMS;
    const int total = WS_ELEMS + WV_ELEMS;
    pack_weights_kernel<<<(total + 255) / 256, 256, 0, stream>>>(W_s, W_v, WpS, WpV);
    si_fused_kernel<true><<<blocks, 64, 0, stream>>>(features, coords, c_noise,
                                                     W_s, W_v, WpS, WpV,
                                                     noise_w, noise_b, outc, outf, n);
  } else {
    si_fused_kernel<false><<<blocks, 64, 0, stream>>>(features, coords, c_noise,
                                                      W_s, W_v, nullptr, nullptr,
                                                      noise_w, noise_b, outc, outf, n);
  }
}
